// PointCloudFeatureExtractor_6107443495554
// MI455X (gfx1250) — matrix operand order repaired and checked by reading, not compiled, not run on hardware
//
#include <hip/hip_runtime.h>
#include <hip/hip_bf16.h>
#include <float.h>

// ---------------------------------------------------------------------------
// Types for CDNA5 WMMA (wave32): bf16x16 = 8 VGPRs packed bf16, floatx8 = acc
// ---------------------------------------------------------------------------
typedef __attribute__((ext_vector_type(16))) __bf16 bf16x16;
typedef __attribute__((ext_vector_type(8)))  __bf16 bf16x8;
typedef __attribute__((ext_vector_type(8)))  float  floatx8;

#define B_BATCH 2
#define V1N 8192
#define V2N 2048
#define V3N 512
#define NN  50

// ---------------------------------------------------------------------------
// KNN: one thread per query point; candidates streamed through LDS tiles,
// next tile prefetched (global_prefetch_b8) while the current one is scanned.
// Maintains ascending top-k (self excluded) with threshold-gated insertion.
// ---------------------------------------------------------------------------
#define KNN_TILE 256
__global__ void knn_kernel(const float* __restrict__ v, int nq, int n, int k,
                           int* __restrict__ idx_out) {
    __shared__ float sx[KNN_TILE], sy[KNN_TILE], sz[KNN_TILE];
    const int b = blockIdx.y;
    const float* vb = v + (size_t)b * n * 3;
    const int q = blockIdx.x * blockDim.x + threadIdx.x;
    const bool active = q < nq;
    float qx = 0.f, qy = 0.f, qz = 0.f;
    if (active) { qx = vb[q*3]; qy = vb[q*3+1]; qz = vb[q*3+2]; }

    float dk[64]; int ik[64];                 // k <= 50
    for (int t = 0; t < k; ++t) { dk[t] = FLT_MAX; ik[t] = 0; }

    for (int base = 0; base < n; base += KNN_TILE) {
        const int tlen = min(KNN_TILE, n - base);
        __syncthreads();
        for (int t = threadIdx.x; t < tlen; t += blockDim.x) {
            sx[t] = vb[(base+t)*3];
            sy[t] = vb[(base+t)*3+1];
            sz[t] = vb[(base+t)*3+2];
        }
        // prefetch next candidate tile (48 x 64B lines) while scanning this one
        if (base + KNN_TILE < n && threadIdx.x < 48) {
            __builtin_prefetch(vb + (size_t)(base + KNN_TILE) * 3 + threadIdx.x * 16, 0, 0);
        }
        __syncthreads();
        if (active) {
            for (int t = 0; t < tlen; ++t) {
                const int j = base + t;
                if (j == q) continue;          // drop self (head of ref top-k)
                const float dx = sx[t]-qx, dy = sy[t]-qy, dz = sz[t]-qz;
                const float d = dx*dx + dy*dy + dz*dz;
                if (d < dk[k-1]) {
                    int p = k - 1;
                    while (p > 0 && dk[p-1] > d) {
                        dk[p] = dk[p-1]; ik[p] = ik[p-1]; --p;
                    }
                    dk[p] = d; ik[p] = j;
                }
            }
        }
    }
    if (active) {
        int* out = idx_out + ((size_t)b * nq + q) * k;
        for (int t = 0; t < k; ++t) out[t] = ik[t];
    }
}

// ---------------------------------------------------------------------------
// conv_surface: fm0[b,p,k] = max_n relu(dnorm . normalize(dir0[:,k]))
// Output is bf16 (consumed only by the first WMMA GEMM).
// ---------------------------------------------------------------------------
__global__ void conv_surface_kernel(const int* __restrict__ idx,
                                    const float* __restrict__ v,
                                    const float* __restrict__ dirs,
                                    __bf16* __restrict__ fm,
                                    int vn, int nn, int kn) {
    __shared__ float dn[NN * 3];
    const int b = blockIdx.y, p = blockIdx.x;
    const float* vb = v + (size_t)b * vn * 3;
    if (threadIdx.x < nn) {
        const int j = idx[((size_t)b * vn + p) * nn + threadIdx.x];
        const float cx = vb[p*3], cy = vb[p*3+1], cz = vb[p*3+2];
        float dx = vb[j*3]-cx, dy = vb[j*3+1]-cy, dz = vb[j*3+2]-cz;
        const float inv = 1.0f / fmaxf(sqrtf(dx*dx + dy*dy + dz*dz), 1e-12f);
        dn[threadIdx.x*3+0] = dx*inv;
        dn[threadIdx.x*3+1] = dy*inv;
        dn[threadIdx.x*3+2] = dz*inv;
    }
    __syncthreads();
    const int c = threadIdx.x;
    if (c < kn) {
        float d0 = dirs[c], d1 = dirs[kn + c], d2 = dirs[2*kn + c];
        const float inv = 1.0f / fmaxf(sqrtf(d0*d0 + d1*d1 + d2*d2), 1e-12f);
        d0 *= inv; d1 *= inv; d2 *= inv;
        float m = 0.f;                        // relu(theta) >= 0
        for (int t = 0; t < nn; ++t) {
            const float th = dn[t*3]*d0 + dn[t*3+1]*d1 + dn[t*3+2]*d2;
            m = fmaxf(m, fmaxf(th, 0.f));
        }
        fm[((size_t)b * vn + p) * kn + c] = (__bf16)m;
    }
}

// ---------------------------------------------------------------------------
// pack_w: W(KxN f32, row-major) -> bf16 pre-swizzled into the WMMA B-operand
// lane layout: tile (nt,kt) holds 32 lanes x 16 contiguous bf16; lane L
// (r=L&15, h=L>>4): element s corresponds to K = kt*32 + 16h + s, N = nt*16+r.
// ---------------------------------------------------------------------------
__global__ void pack_w_kernel(const float* __restrict__ W, __bf16* __restrict__ Wp,
                              int K, int N) {
    const int flat = blockIdx.x * blockDim.x + threadIdx.x;
    if (flat >= K * N) return;
    const int s = flat & 15;
    const int lane = (flat >> 4) & 31;
    const int rest = flat >> 9;
    const int KT = K >> 5;
    const int kt = rest % KT, nt = rest / KT;
    const int k = kt * 32 + (lane >> 4) * 16 + s;
    const int n = nt * 16 + (lane & 15);
    Wp[flat] = (__bf16)W[(size_t)k * N + n];
}

// ---------------------------------------------------------------------------
// WMMA GEMM: out[b] = A[b](MxK,bf16) @ W(KxN) + bias, f32 accumulate.
// Block = 256 = 8 waves; wave tile = 16 rows x 32 cols (two 16x16 subtiles
// sharing one A fragment -> 2 wmma per 6 b128 loads per lane per k-step).
// A lane layout (16-bit A 16x32): lane r/h, two contiguous 8-elem runs at
// K = 16h and K = 16h+8.  B fragment: 32 contiguous bytes from packed Wp.
// C/D f32 16x16: VGPR j -> row = j + 8h, col = r.
// ---------------------------------------------------------------------------
__global__ void gemm_wmma_kernel(const __bf16* __restrict__ A,
                                 const __bf16* __restrict__ Wp,
                                 const float* __restrict__ bias,
                                 float* __restrict__ out,
                                 int M, int K, int N) {
    const int b = blockIdx.z;
    const __bf16* Ab = A + (size_t)b * M * K;
    float* Ob = out + (size_t)b * M * N;
    const int lane = threadIdx.x & 31;
    const int wave = threadIdx.x >> 5;
    const int mrow = blockIdx.y * 32 + (wave >> 2) * 16;
    const int ncol = blockIdx.x * 128 + (wave & 3) * 32;   // subtiles at ncol, ncol+16
    const int r = lane & 15, h = lane >> 4;
    const int KT = K >> 5;
    const int nt0 = ncol >> 4;

    const __bf16* arow = Ab + (size_t)(mrow + r) * K + 16 * h;
    const __bf16* wp0 = Wp + (((size_t)nt0 * KT) * 32 + lane) * 16;
    const __bf16* wp1 = Wp + (((size_t)(nt0 + 1) * KT) * 32 + lane) * 16;

    floatx8 acc0 = {}, acc1 = {};
    for (int kt = 0; kt < KT; ++kt) {
        const bf16x8 alo = *(const bf16x8*)(arow + kt * 32);
        const bf16x8 ahi = *(const bf16x8*)(arow + kt * 32 + 8);
        const bf16x16 av = __builtin_shufflevector(alo, ahi,
            0,1,2,3,4,5,6,7,8,9,10,11,12,13,14,15);
        const bf16x16 bv0 = *(const bf16x16*)(wp0 + (size_t)kt * 512);
        const bf16x16 bv1 = *(const bf16x16*)(wp1 + (size_t)kt * 512);
        acc0 = __builtin_amdgcn_wmma_f32_16x16x32_bf16(
                   false, av, false, bv0, (short)0, acc0, false, false);
        acc1 = __builtin_amdgcn_wmma_f32_16x16x32_bf16(
                   false, av, false, bv1, (short)0, acc1, false, false);
    }
    #pragma unroll
    for (int j = 0; j < 8; ++j) {
        const int row = mrow + h * 8 + j;
        Ob[(size_t)row * N + ncol + r]      = acc0[j] + bias[ncol + r];
        Ob[(size_t)row * N + ncol + 16 + r] = acc1[j] + bias[ncol + 16 + r];
    }
}

// ---------------------------------------------------------------------------
// conv_layer aggregation (templated output type: f32, or bf16 when the only
// consumer is the next WMMA GEMM):
// fm[b,p,c] = relu?( fout[b,p,c] + max_n( relu(dnorm.sdir_c) *
//                                         fout[b,idx[n], out_c + c] ) )
// ---------------------------------------------------------------------------
template <typename OutT>
__global__ void conv_agg_kernel(const int* __restrict__ idx,
                                const float* __restrict__ v,
                                const float* __restrict__ fout,
                                const float* __restrict__ dirs,
                                OutT* __restrict__ fm,
                                int vn, int nn, int out_c, int relu) {
    __shared__ float dn[NN * 3];
    __shared__ int   nid[NN];
    const int b = blockIdx.y, p = blockIdx.x;
    const float* vb = v + (size_t)b * vn * 3;
    const float* fb = fout + (size_t)b * vn * 2 * out_c;
    if (threadIdx.x < nn) {
        const int j = idx[((size_t)b * vn + p) * nn + threadIdx.x];
        nid[threadIdx.x] = j;
        const float cx = vb[p*3], cy = vb[p*3+1], cz = vb[p*3+2];
        float dx = vb[j*3]-cx, dy = vb[j*3+1]-cy, dz = vb[j*3+2]-cz;
        const float inv = 1.0f / fmaxf(sqrtf(dx*dx + dy*dy + dz*dz), 1e-12f);
        dn[threadIdx.x*3+0] = dx*inv;
        dn[threadIdx.x*3+1] = dy*inv;
        dn[threadIdx.x*3+2] = dz*inv;
    }
    __syncthreads();
    for (int c = threadIdx.x; c < out_c; c += blockDim.x) {
        float d0 = dirs[c], d1 = dirs[out_c + c], d2 = dirs[2*out_c + c];
        const float inv = 1.0f / fmaxf(sqrtf(d0*d0 + d1*d1 + d2*d2), 1e-12f);
        d0 *= inv; d1 *= inv; d2 *= inv;
        const float fc = fb[(size_t)p * 2 * out_c + c];
        float m = -FLT_MAX;
        for (int t = 0; t < nn; ++t) {
            const float th = fmaxf(dn[t*3]*d0 + dn[t*3+1]*d1 + dn[t*3+2]*d2, 0.f);
            const float fs = fb[(size_t)nid[t] * 2 * out_c + out_c + c];
            m = fmaxf(m, th * fs);
        }
        float res = fc + m;
        if (relu) res = fmaxf(res, 0.f);
        fm[((size_t)b * vn + p) * out_c + c] = (OutT)res;
    }
}

// ---------------------------------------------------------------------------
// pool: max over 4 KNN neighbors, first pn rows only; bf16 out (GEMM input).
// ---------------------------------------------------------------------------
__global__ void pool_gather_kernel(const float* __restrict__ fmv,
                                   const int* __restrict__ idx4,
                                   __bf16* __restrict__ out,
                                   int vn_in, int pn, int c, int nn) {
    const size_t tid = (size_t)blockIdx.x * blockDim.x + threadIdx.x;
    const size_t total = (size_t)B_BATCH * pn * c;
    if (tid >= total) return;
    const int ch = (int)(tid % c);
    const size_t pc = tid / c;
    const int p = (int)(pc % pn);
    const int b = (int)(pc / pn);
    const float* fb = fmv + (size_t)b * vn_in * c;
    const int* ib = idx4 + ((size_t)b * pn + p) * nn;
    float m = -FLT_MAX;
    for (int t = 0; t < nn; ++t) m = fmaxf(m, fb[(size_t)ib[t] * c + ch]);
    out[tid] = (__bf16)m;
}

__global__ void vcopy_kernel(const float* __restrict__ vin,
                             float* __restrict__ vout, int vn_in, int pn) {
    const int t = blockIdx.x * blockDim.x + threadIdx.x;
    if (t >= B_BATCH * pn * 3) return;
    const int b = t / (pn * 3);
    const int rem = t % (pn * 3);
    vout[t] = vin[(size_t)b * vn_in * 3 + rem];
}

// ---------------------------------------------------------------------------
// feat[b,c] = max_v fm4[b,v,c] ; then out = feat @ fc_w + fc_b (tiny)
// ---------------------------------------------------------------------------
__global__ void featmax_kernel(const float* __restrict__ fm4,
                               float* __restrict__ feat, int vn, int c) {
    const int t = blockIdx.x * blockDim.x + threadIdx.x;
    if (t >= B_BATCH * c) return;
    const int b = t / c, ch = t % c;
    const float* fb = fm4 + (size_t)b * vn * c;
    float m = -FLT_MAX;
    for (int vv = 0; vv < vn; ++vv) m = fmaxf(m, fb[(size_t)vv * c + ch]);
    feat[t] = m;
}

__global__ void fc_kernel(const float* __restrict__ feat,
                          const float* __restrict__ w,
                          const float* __restrict__ bias,
                          float* __restrict__ out) {
    const int j = threadIdx.x;      // 256
    const int b = blockIdx.x;       // 2
    float s = bias[j];
    for (int c = 0; c < 1024; ++c) s += feat[b*1024 + c] * w[c*256 + j];
    out[b*256 + j] = s;
}

// ---------------------------------------------------------------------------
// Host orchestration
// ---------------------------------------------------------------------------
static inline size_t alignUp(size_t x, size_t a) { return (x + a - 1) & ~(a - 1); }

extern "C" void kernel_launch(void* const* d_in, const int* in_sizes, int n_in,
                              void* d_out, int out_size, void* d_ws, size_t ws_size,
                              hipStream_t stream) {
    const float* vertices = (const float*)d_in[0];
    const float* dir0 = (const float*)d_in[1];
    const float* w1 = (const float*)d_in[2];  const float* b1 = (const float*)d_in[3];
    const float* d1 = (const float*)d_in[4];
    const float* w2 = (const float*)d_in[5];  const float* b2 = (const float*)d_in[6];
    const float* d2 = (const float*)d_in[7];
    const float* w3 = (const float*)d_in[8];  const float* b3 = (const float*)d_in[9];
    const float* d3 = (const float*)d_in[10];
    const float* w4 = (const float*)d_in[11]; const float* b4 = (const float*)d_in[12];
    const float* d4 = (const float*)d_in[13];
    const float* fcw = (const float*)d_in[14]; const float* fcb = (const float*)d_in[15];

    char* ws = (char*)d_ws;
    size_t off = 0;
    auto alloc = [&](size_t bytes) -> void* {
        void* p = ws + off;
        off = alignUp(off + bytes, 256);
        return p;
    };

    int*    idxA  = (int*)   alloc((size_t)B_BATCH*V1N*NN*4);
    __bf16* fm0b  = (__bf16*)alloc((size_t)B_BATCH*V1N*32*2);
    float*  fout1 = (float*) alloc((size_t)B_BATCH*V1N*128*4);
    float*  fm1   = (float*) alloc((size_t)B_BATCH*V1N*64*4);
    int*    idx4A = (int*)   alloc((size_t)B_BATCH*V2N*4*4);
    __bf16* fm1p  = (__bf16*)alloc((size_t)B_BATCH*V2N*64*2);
    float*  v2b   = (float*) alloc((size_t)B_BATCH*V2N*3*4);
    int*    idxB  = (int*)   alloc((size_t)B_BATCH*V2N*NN*4);
    float*  fout2 = (float*) alloc((size_t)B_BATCH*V2N*256*4);
    __bf16* fm2b  = (__bf16*)alloc((size_t)B_BATCH*V2N*128*2);
    float*  fout3 = (float*) alloc((size_t)B_BATCH*V2N*512*4);
    float*  fm3   = (float*) alloc((size_t)B_BATCH*V2N*256*4);
    int*    idx4B = (int*)   alloc((size_t)B_BATCH*V3N*4*4);
    __bf16* fm3p  = (__bf16*)alloc((size_t)B_BATCH*V3N*256*2);
    float*  v3b   = (float*) alloc((size_t)B_BATCH*V3N*3*4);
    int*    idxC  = (int*)   alloc((size_t)B_BATCH*V3N*NN*4);
    float*  fout4 = (float*) alloc((size_t)B_BATCH*V3N*2048*4);
    float*  fm4   = (float*) alloc((size_t)B_BATCH*V3N*1024*4);
    float*  feat  = (float*) alloc((size_t)B_BATCH*1024*4);
    __bf16* wp1   = (__bf16*)alloc((size_t)32*128*2);
    __bf16* wp2   = (__bf16*)alloc((size_t)64*256*2);
    __bf16* wp3   = (__bf16*)alloc((size_t)128*512*2);
    __bf16* wp4   = (__bf16*)alloc((size_t)256*2048*2);
    (void)ws_size; (void)n_in; (void)in_sizes; (void)out_size;

    auto g1 = [](size_t n) { return dim3((unsigned)((n + 255) / 256)); };

    // ---- One-time weight repacks (f32 -> bf16, WMMA B-operand layout) ----
    pack_w_kernel<<<g1(32*128),   256, 0, stream>>>(w1, wp1, 32, 128);
    pack_w_kernel<<<g1(64*256),   256, 0, stream>>>(w2, wp2, 64, 256);
    pack_w_kernel<<<g1(128*512),  256, 0, stream>>>(w3, wp3, 128, 512);
    pack_w_kernel<<<g1(256*2048), 256, 0, stream>>>(w4, wp4, 256, 2048);

    // ---- Stage 1 (vn = 8192) ----
    knn_kernel<<<dim3((V1N+255)/256, B_BATCH), 256, 0, stream>>>(vertices, V1N, V1N, NN, idxA);
    conv_surface_kernel<<<dim3(V1N, B_BATCH), 64, 0, stream>>>(idxA, vertices, dir0, fm0b, V1N, NN, 32);
    gemm_wmma_kernel<<<dim3(128/128, V1N/32, B_BATCH), 256, 0, stream>>>(fm0b, wp1, b1, fout1, V1N, 32, 128);
    conv_agg_kernel<float><<<dim3(V1N, B_BATCH), 64, 0, stream>>>(idxA, vertices, fout1, d1, fm1, V1N, NN, 64, 1);
    knn_kernel<<<dim3((V2N+255)/256, B_BATCH), 256, 0, stream>>>(vertices, V2N, V1N, 4, idx4A);
    pool_gather_kernel<<<g1((size_t)B_BATCH*V2N*64), 256, 0, stream>>>(fm1, idx4A, fm1p, V1N, V2N, 64, 4);
    vcopy_kernel<<<g1((size_t)B_BATCH*V2N*3), 256, 0, stream>>>(vertices, v2b, V1N, V2N);

    // ---- Stage 2 (vn = 2048) ----
    knn_kernel<<<dim3((V2N+255)/256, B_BATCH), 256, 0, stream>>>(v2b, V2N, V2N, NN, idxB);
    gemm_wmma_kernel<<<dim3(256/128, V2N/32, B_BATCH), 256, 0, stream>>>(fm1p, wp2, b2, fout2, V2N, 64, 256);
    conv_agg_kernel<__bf16><<<dim3(V2N, B_BATCH), 128, 0, stream>>>(idxB, v2b, fout2, d2, fm2b, V2N, NN, 128, 1);
    gemm_wmma_kernel<<<dim3(512/128, V2N/32, B_BATCH), 256, 0, stream>>>(fm2b, wp3, b3, fout3, V2N, 128, 512);
    conv_agg_kernel<float><<<dim3(V2N, B_BATCH), 256, 0, stream>>>(idxB, v2b, fout3, d3, fm3, V2N, NN, 256, 1);
    knn_kernel<<<dim3((V3N+255)/256, B_BATCH), 256, 0, stream>>>(v2b, V3N, V2N, 4, idx4B);
    pool_gather_kernel<<<g1((size_t)B_BATCH*V3N*256), 256, 0, stream>>>(fm3, idx4B, fm3p, V2N, V3N, 256, 4);
    vcopy_kernel<<<g1((size_t)B_BATCH*V3N*3), 256, 0, stream>>>(v2b, v3b, V2N, V3N);

    // ---- Stage 3 (vn = 512) ----
    knn_kernel<<<dim3((V3N+255)/256, B_BATCH), 256, 0, stream>>>(v3b, V3N, V3N, NN, idxC);
    gemm_wmma_kernel<<<dim3(2048/128, V3N/32, B_BATCH), 256, 0, stream>>>(fm3p, wp4, b4, fout4, V3N, 256, 2048);
    conv_agg_kernel<float><<<dim3(V3N, B_BATCH), 256, 0, stream>>>(idxC, v3b, fout4, d4, fm4, V3N, NN, 1024, 0);

    // ---- Head ----
    featmax_kernel<<<g1((size_t)B_BATCH*1024), 256, 0, stream>>>(fm4, feat, V3N, 1024);
    fc_kernel<<<B_BATCH, 256, 0, stream>>>(feat, fcw, fcb, (float*)d_out);
}
